// Model_36532991819820
// MI455X (gfx1250) — matrix operand order repaired and checked by reading, compiled, not run on hardware
//
#include <hip/hip_runtime.h>
#include <hip/hip_bf16.h>

// ---- problem constants (match reference) ----
constexpr int NB  = 4096;   // batch
constexpr int NS  = 50;     // sequence
constexpr int NF  = 28;     // input feature dim (1 id + 27 feats)
constexpr int NE  = 50;     // embed dim
constexpr int ND  = 77;     // per-half feature dim
constexpr int ND2 = 154;    // h dim
constexpr int NH  = 616;    // 4*ND2
constexpr int NHP = 640;    // NH padded to 32 for WMMA K / 16 for N tiles
constexpr int NV  = 50000;  // vocab
constexpr int NVP = 50176;  // NV padded to 128-col tiles (392*128)
constexpr int KP  = 160;    // ND2 padded to multiple of 32 for WMMA K
constexpr int DP  = 160;    // ND2 output cols padded to 16

typedef __attribute__((ext_vector_type(16))) __bf16          v16bf;
typedef __attribute__((ext_vector_type(16))) unsigned short  v16u;
typedef __attribute__((ext_vector_type(8)))  float           v8f;
typedef __attribute__((ext_vector_type(4)))  unsigned int    uiv4;
typedef __attribute__((ext_vector_type(8)))  int             iv8;
typedef __attribute__((ext_vector_type(4)))  int             iv4;

__device__ __forceinline__ unsigned short f2bf(float f) {
  unsigned u = __builtin_bit_cast(unsigned, f);
  u += 0x7FFFu + ((u >> 16) & 1u);           // round-to-nearest-even
  return (unsigned short)(u >> 16);
}

#if defined(__has_builtin)
#if __has_builtin(__builtin_amdgcn_tensor_load_to_lds) && __has_builtin(__builtin_amdgcn_s_wait_tensorcnt)
#define HAVE_TDM 1
#endif
#endif
#ifndef HAVE_TDM
#define HAVE_TDM 0
#endif

#if HAVE_TDM
// TDM strided-row DMA: `rows` rows of `row_elems` bf16 from global (row stride
// row_elems) into LDS rows of stride lds_row_elems. Iterate-mode D# per ISA §8:
// per-iteration tile = 1 x row_elems, global/lds address increments per row.
// Must be executed by exactly one wave; waits TENSORcnt==0 before returning.
// This toolchain exposes the 6-operand builtin: the 5th (int32x8) operand is
// zero-filled (only D# groups 0-3 are defined by the ISA), cpol = 0.
__device__ __forceinline__ void tdm_load_rows(unsigned lds_byte_addr, const void* gptr,
                                              int rows, int row_elems, int lds_row_elems)
{
  unsigned long long ga = (unsigned long long)gptr;
  uiv4 g0; iv8 g1; iv4 g2, g3; iv8 gz = {};
  g0[0] = 1u;                                            // count=1 (valid D#)
  g0[1] = lds_byte_addr;                                 // lds_addr
  g0[2] = (unsigned)(ga & 0xffffffffu);                  // global_addr lo
  g0[3] = (unsigned)((ga >> 32) & 0x1ffffffu) | (2u << 30); // addr[56:32] | type=2
  g1[0] = (int)((1u << 16) | (1u << 19));                // data_size=2B, iterate_enable
  g1[1] = (int)((unsigned)(row_elems & 0xffff) << 16);   // tensor_dim0 lo16
  g1[2] = (int)(1u << 16);                               // tensor_dim1 = 1
  g1[3] = (int)((unsigned)(row_elems & 0xffff) << 16);   // tile_dim0
  g1[4] = 1;                                             // tile_dim1 = 1
  g1[5] = row_elems;                                     // tensor_dim0_stride
  g1[6] = 0; g1[7] = 0;
  g2[0] = 0;                                             // tensor_dim2
  g2[1] = lds_row_elems;                                 // lds_addr_increment (elems)
  g2[2] = row_elems;                                     // global_addr_increment lo
  g2[3] = (int)((unsigned)(rows - 1) << 16);             // iterate_count
  g3[0] = 0; g3[1] = 0; g3[2] = 0; g3[3] = 0;
  __builtin_amdgcn_tensor_load_to_lds(g0, g1, g2, g3, gz, 0);
  __builtin_amdgcn_s_wait_tensorcnt(0);
}
#endif

// ---------------------------------------------------------------------------
// K0: one-shot weight conversion f32 -> bf16 with zero padding.
//  W1bf: [6][NHP][KP]   from res_w1 [6][616][154]
//  W2bf: [6][DP][NHP]   from res_w2 [6][154][616]
//  Wfbf: [NVP][KP]      from fc_w   [NV][154]
// ---------------------------------------------------------------------------
__global__ void k_convert_weights(const float* __restrict__ res_w1,
                                  const float* __restrict__ res_w2,
                                  const float* __restrict__ fc_w,
                                  unsigned short* __restrict__ W1bf,
                                  unsigned short* __restrict__ W2bf,
                                  unsigned short* __restrict__ Wfbf)
{
  const long long n1 = 6LL * NHP * KP;
  const long long n2 = 6LL * DP * NHP;
  const long long n3 = (long long)NVP * KP;
  const long long tot = n1 + n2 + n3;
  long long stride = (long long)gridDim.x * blockDim.x;
  for (long long idx = (long long)blockIdx.x * blockDim.x + threadIdx.x; idx < tot; idx += stride) {
    if (idx < n1) {
      long long i = idx / (NHP * KP), rem = idx % (NHP * KP);
      int n = (int)(rem / KP), k = (int)(rem % KP);
      float v = (n < NH && k < ND2) ? res_w1[(i * NH + n) * ND2 + k] : 0.f;
      W1bf[idx] = f2bf(v);
    } else if (idx < n1 + n2) {
      long long j = idx - n1;
      long long i = j / (DP * NHP), rem = j % (DP * NHP);
      int d = (int)(rem / NHP), k = (int)(rem % NHP);
      float v = (d < ND2 && k < NH) ? res_w2[(i * ND2 + d) * NH + k] : 0.f;
      W2bf[j] = f2bf(v);
    } else {
      long long j = idx - n1 - n2;
      int v = (int)(j / KP), k = (int)(j % KP);
      float w = (v < NV && k < ND2) ? fc_w[(long long)v * ND2 + k] : 0.f;
      Wfbf[j] = f2bf(w);
    }
  }
}

// ---------------------------------------------------------------------------
// K1: embedding gather + masked mean + next-item + SE gate -> h0 (NB x ND2 f32)
// ---------------------------------------------------------------------------
__global__ void k_build_h(const float* __restrict__ input, const int* __restrict__ mask,
                          const float* __restrict__ embed,
                          const float* __restrict__ se_w1, const float* __restrict__ se_b1,
                          const float* __restrict__ se_w2, const float* __restrict__ se_b2,
                          float* __restrict__ h0)
{
  __shared__ int   ids_s[NS];
  __shared__ int   msk[NS];
  __shared__ float hrow[ND2];
  __shared__ float means[6];
  __shared__ float wse[6];
  __shared__ float leninv;

  const int b = blockIdx.x;
  const int t = threadIdx.x;
  const float* inp = input + (size_t)b * NS * NF;

  if (t < NS) { ids_s[t] = (int)inp[t * NF]; msk[t] = mask[b * NS + t]; }
  __syncthreads();
  if (t == 0) { int L = 0; for (int s = 0; s < NS; ++s) L += msk[s]; leninv = 1.0f / (float)L; }
  __syncthreads();

  if (t < ND2) {
    float v;
    if (t < ND) {
      float acc = 0.f;
      for (int s = 0; s < NS; ++s) {
        if (msk[s]) {
          float x = (t < NE) ? embed[(size_t)ids_s[s] * NE + t]
                             : inp[s * NF + 1 + (t - NE)];
          acc += x;
        }
      }
      v = acc * leninv;
    } else {
      int d2 = t - ND;
      v = (d2 < NE) ? embed[(size_t)ids_s[NS - 1] * NE + d2]
                    : inp[(NS - 1) * NF + 1 + (d2 - NE)];
    }
    hrow[t] = v;
  }
  __syncthreads();

  if (t < 6) {
    const int lo[6] = {0, 50, 56, 77, 127, 133};
    const int hi[6] = {50, 56, 77, 127, 133, 154};
    float s = 0.f;
    for (int d = lo[t]; d < hi[t]; ++d) s += hrow[d];
    means[t] = s / (float)(hi[t] - lo[t]);
  }
  __syncthreads();

  if (t == 0) {
    float hid[2];
    for (int j = 0; j < 2; ++j) {
      float a = se_b1[j];
      for (int i = 0; i < 6; ++i) a += means[i] * se_w1[j * 6 + i];
      hid[j] = fmaxf(a, 0.f);
    }
    for (int o = 0; o < 6; ++o) {
      float a = se_b2[o];
      for (int j = 0; j < 2; ++j) a += hid[j] * se_w2[o * 2 + j];
      wse[o] = 1.0f / (1.0f + expf(-a));
    }
  }
  __syncthreads();

  if (t < ND2) {
    int sgi = (t < 50) ? 0 : (t < 56) ? 1 : (t < 77) ? 2 : (t < 127) ? 3 : (t < 133) ? 4 : 5;
    h0[(size_t)b * ND2 + t] = hrow[t] * wse[sgi];
  }
}

// ---------------------------------------------------------------------------
// K2: 6 residual MLP blocks via WMMA bf16. 16 batch rows / workgroup, 8 waves.
// Master h kept f32 in LDS; A operands staged bf16 (stride 168 = conflict-free);
// B fragments stream from L2 as 32B/lane contiguous loads (k = 16*g + e).
// ---------------------------------------------------------------------------
#define ASTR2 168   // abf row stride (ushorts): 336B, conflict-free, 16B aligned
#define YSTR  648   // ybf row stride (ushorts): 1296B = 324 dwords, conflict-free

__global__ void k_resblocks_wmma(const unsigned short* __restrict__ W1bf,
                                 const float* __restrict__ res_b1,
                                 const unsigned short* __restrict__ W2bf,
                                 const float* __restrict__ res_b2,
                                 const float* __restrict__ h0,
                                 unsigned short* __restrict__ hbf)
{
  __shared__ float          hs[16 * ND2];     // 9.9 KB  master h (f32)
  __shared__ unsigned short abf[16 * ASTR2];  // 5.25 KB A for GEMM1
  __shared__ unsigned short ybf[16 * YSTR];   // 20.25 KB A for GEMM2

  const int t = threadIdx.x;                  // 256 threads, 8 waves
  const int bbase = blockIdx.x * 16;
  const int lane = t & 31, wv = t >> 5;
  const int g = lane >> 4, nn = lane & 15;

  for (int idx = t; idx < 16 * ND2; idx += 256)
    hs[idx] = h0[(size_t)bbase * ND2 + idx];

  for (int i = 0; i < 6; ++i) {
    __syncthreads();
    // stage h -> bf16 A tile (K zero-padded to KP)
    for (int idx = t; idx < 16 * KP; idx += 256) {
      int rr = idx / KP, k = idx - rr * KP;
      abf[rr * ASTR2 + k] = (k < ND2) ? f2bf(hs[rr * ND2 + k]) : (unsigned short)0;
    }
    __syncthreads();

    // GEMM1: y(16 x NHP) = h(16 x KP) @ W1^T ; 40 n-subtiles, 5 per wave
    const unsigned short* W1i = W1bf + (size_t)i * NHP * KP;
    for (int j = 0; j < 5; ++j) {
      const int nbase = (wv + 8 * j) * 16;
      v8f c = {};
      for (int kb = 0; kb < KP; kb += 32) {
        v16u au;
        #pragma unroll
        for (int e = 0; e < 16; ++e) {
          int ka = kb + 16 * g + e;
          au[e] = abf[nn * ASTR2 + ka];
        }
        v16u bu = *(const v16u*)(W1i + (size_t)(nbase + nn) * KP + kb + 16 * g);
        c = __builtin_amdgcn_wmma_f32_16x16x32_bf16(false, __builtin_bit_cast(v16bf, au),
                                                    false, __builtin_bit_cast(v16bf, bu),
                                                    (short)0, c, false, false);
      }
      const int ncol = nbase + nn;
      const float bias = (ncol < NH) ? res_b1[i * NH + ncol] : 0.f;
      #pragma unroll
      for (int rr = 0; rr < 8; ++rr) {
        int m = rr + 8 * g;
        ybf[m * YSTR + ncol] = f2bf(fmaxf(c[rr] + bias, 0.f));
      }
    }
    __syncthreads();

    // GEMM2: z(16 x DP) = y(16 x NHP) @ W2^T ; 10 d-subtiles (waves 0..7 + 0,1)
    const unsigned short* W2i = W2bf + (size_t)i * DP * NHP;
    for (int j = 0; j < 2; ++j) {
      const int ds = wv + 8 * j;
      if (ds < 10) {                            // wave-uniform branch
        const int dbase = ds * 16;
        v8f c = {};
        for (int kb = 0; kb < NHP; kb += 32) {
          v16u au;
          #pragma unroll
          for (int e = 0; e < 16; ++e) {
            int ka = kb + 16 * g + e;
            au[e] = ybf[nn * YSTR + ka];
          }
          v16u bu = *(const v16u*)(W2i + (size_t)(dbase + nn) * NHP + kb + 16 * g);
          c = __builtin_amdgcn_wmma_f32_16x16x32_bf16(false, __builtin_bit_cast(v16bf, au),
                                                      false, __builtin_bit_cast(v16bf, bu),
                                                      (short)0, c, false, false);
        }
        const int d = dbase + nn;
        if (d < ND2) {
          const float bias = res_b2[i * ND2 + d];
          #pragma unroll
          for (int rr = 0; rr < 8; ++rr) {
            int m = rr + 8 * g;
            hs[m * ND2 + d] += fmaxf(c[rr] + bias, 0.f);   // residual add (f32)
          }
        }
      }
    }
  }
  __syncthreads();

  for (int idx = t; idx < 16 * KP; idx += 256) {   // bf16 h for FC, zero-padded
    int rr = idx / KP, k = idx - rr * KP;
    float v = (k < ND2) ? hs[rr * ND2 + k] : 0.f;
    hbf[(size_t)(bbase + rr) * KP + k] = f2bf(v);
  }
}

// ---------------------------------------------------------------------------
// K3: logits = h @ fc_w^T + fc_b via v_wmma_f32_16x16x32_bf16.
// 128-col tile staged in LDS (TDM strided-row DMA when available), reused over
// 16 M-tiles. 4 waves x 2 n-subtiles.
// ---------------------------------------------------------------------------
#define BSTR 168
#define ASTR 168

__global__ void k_fc_wmma(const unsigned short* __restrict__ hbf,
                          const unsigned short* __restrict__ Wfbf,
                          const float* __restrict__ fc_b,
                          float* __restrict__ out)
{
  __shared__ unsigned short Bsh[128 * BSTR];  // 42 KB weight tile
  __shared__ unsigned short Ash[16 * ASTR];   // 5.25 KB activation tile

  const int t     = threadIdx.x;              // 128 threads, 4 waves
  const int vbase = blockIdx.x * 128;
  const int lane  = t & 31, wv = t >> 5;
  const int g     = lane >> 4, nn = lane & 15;
  const int cbase = wv * 32;
  const int v0    = vbase + cbase + nn;
  const int v1    = v0 + 16;
  const float bias0 = (v0 < NV) ? fc_b[v0] : 0.f;
  const float bias1 = (v1 < NV) ? fc_b[v1] : 0.f;

#if HAVE_TDM
  if (wv == 0)
    tdm_load_rows((unsigned)(unsigned long long)(void*)Bsh,
                  Wfbf + (size_t)vbase * KP, 128, KP, BSTR);
#else
  for (int idx = t; idx < 128 * KP; idx += 128) {
    int c = idx / KP, k = idx - c * KP;
    Bsh[c * BSTR + k] = Wfbf[(size_t)(vbase + c) * KP + k];
  }
#endif

  for (int mt = 0; mt < 16; ++mt) {
    const int rbase = (blockIdx.y * 16 + mt) * 16;
    __syncthreads();                                   // Ash free / Bsh ready
#if HAVE_TDM
    if (wv == 0)
      tdm_load_rows((unsigned)(unsigned long long)(void*)Ash,
                    hbf + (size_t)rbase * KP, 16, KP, ASTR);
#else
    for (int idx = t; idx < 16 * KP; idx += 128) {
      int rr = idx / KP, k = idx - rr * KP;
      Ash[rr * ASTR + k] = hbf[(size_t)(rbase + rr) * KP + k];
    }
#endif
    __syncthreads();

    v8f c0 = {}; v8f c1 = {};
    for (int kb = 0; kb < KP; kb += 32) {
      v16u au, bu0, bu1;
      #pragma unroll
      for (int e = 0; e < 16; ++e) {
        int ka = kb + 16 * g + e;
        au[e] = Ash[nn * ASTR + ka];
      }
      #pragma unroll
      for (int e = 0; e < 16; ++e) {
        int kbv = kb + 16 * g + e;
        bu0[e] = Bsh[(cbase + nn) * BSTR + kbv];
        bu1[e] = Bsh[(cbase + 16 + nn) * BSTR + kbv];
      }
      v16bf a = __builtin_bit_cast(v16bf, au);
      c0 = __builtin_amdgcn_wmma_f32_16x16x32_bf16(false, a, false, __builtin_bit_cast(v16bf, bu0),
                                                   (short)0, c0, false, false);
      c1 = __builtin_amdgcn_wmma_f32_16x16x32_bf16(false, a, false, __builtin_bit_cast(v16bf, bu1),
                                                   (short)0, c1, false, false);
    }

    #pragma unroll
    for (int rr = 0; rr < 8; ++rr) {
      int row = rbase + 8 * g + rr;
      if (v0 < NV) out[(size_t)row * NV + v0] = c0[rr] + bias0;
      if (v1 < NV) out[(size_t)row * NV + v1] = c1[rr] + bias1;
    }
  }
}

// ---------------------------------------------------------------------------
// K4: in-place row softmax over NV columns.
// ---------------------------------------------------------------------------
__global__ void k_softmax(float* __restrict__ out)
{
  __shared__ float red[512];
  const int t = threadIdx.x;
  float* row = out + (size_t)blockIdx.x * NV;

  float mx = -3.4e38f;
  for (int i = t; i < NV; i += 512) mx = fmaxf(mx, row[i]);
  red[t] = mx; __syncthreads();
  for (int s = 256; s > 0; s >>= 1) { if (t < s) red[t] = fmaxf(red[t], red[t + s]); __syncthreads(); }
  const float rmax = red[0];
  __syncthreads();

  float sum = 0.f;
  for (int i = t; i < NV; i += 512) sum += __expf(row[i] - rmax);
  red[t] = sum; __syncthreads();
  for (int s = 256; s > 0; s >>= 1) { if (t < s) red[t] += red[t + s]; __syncthreads(); }
  const float inv = 1.0f / red[0];
  __syncthreads();

  for (int i = t; i < NV; i += 512) row[i] = __expf(row[i] - rmax) * inv;
}

// ---------------------------------------------------------------------------
extern "C" void kernel_launch(void* const* d_in, const int* in_sizes, int n_in,
                              void* d_out, int out_size, void* d_ws, size_t ws_size,
                              hipStream_t stream)
{
  (void)in_sizes; (void)n_in; (void)out_size; (void)ws_size;
  const float* input  = (const float*)d_in[0];
  const int*   mask   = (const int*)  d_in[1];
  const float* embed  = (const float*)d_in[2];
  const float* se_w1  = (const float*)d_in[3];
  const float* se_b1  = (const float*)d_in[4];
  const float* se_w2  = (const float*)d_in[5];
  const float* se_b2  = (const float*)d_in[6];
  const float* res_w1 = (const float*)d_in[7];
  const float* res_b1 = (const float*)d_in[8];
  const float* res_w2 = (const float*)d_in[9];
  const float* res_b2 = (const float*)d_in[10];
  const float* fc_w   = (const float*)d_in[11];
  const float* fc_b   = (const float*)d_in[12];
  float* out = (float*)d_out;

  // workspace layout (all 256B aligned)
  char* ws = (char*)d_ws;
  float*          h0   = (float*)ws;                                  // NB*ND2 f32
  unsigned short* hbf  = (unsigned short*)(ws + 2523136);             // NB*KP bf16
  unsigned short* W1bf = (unsigned short*)(ws + 3833856);             // 6*NHP*KP
  unsigned short* W2bf = (unsigned short*)(ws + 5062656);             // 6*DP*NHP
  unsigned short* Wfbf = (unsigned short*)(ws + 6291456);             // NVP*KP

  k_convert_weights<<<2048, 256, 0, stream>>>(res_w1, res_w2, fc_w, W1bf, W2bf, Wfbf);
  k_build_h        <<<NB,   256, 0, stream>>>(input, mask, embed, se_w1, se_b1, se_w2, se_b2, h0);
  k_resblocks_wmma <<<NB/16,256, 0, stream>>>(W1bf, res_b1, W2bf, res_b2, h0, hbf);
  dim3 g3((NV + 127) / 128, 16);
  k_fc_wmma        <<<g3,   128, 0, stream>>>(hbf, Wfbf, fc_b, out);
  k_softmax        <<<NB,   512, 0, stream>>>(out);
}
